// UniPoolModel_32899449487928
// MI455X (gfx1250) — matrix operand order repaired and checked by reading, not compiled, not run on hardware
//
#include <hip/hip_runtime.h>
#include <hip/hip_bf16.h>

// Problem constants (from reference): B=8, S=2048, D=1024, FF=4096, E=16, L=4
#define T_TOK   16384
#define DIM     1024
#define FFD     4096
#define NEXP    16
#define NLAY    4
#define TILE_M  16
#define FFB     128
#define MAX_TILES (T_TOK / TILE_M + NEXP)   // 1040

typedef __attribute__((ext_vector_type(16))) __bf16 v16bf;
typedef __attribute__((ext_vector_type(8)))  float  v8f;

// ---- fp32 -> bf16 (round to nearest even) helpers ----
static __device__ __forceinline__ unsigned pk2bf(float a, float b) {
  unsigned ua = __builtin_bit_cast(unsigned, a);
  unsigned ub = __builtin_bit_cast(unsigned, b);
  ua = (ua + 0x7FFFu + ((ua >> 16) & 1u)) >> 16;
  ub = (ub + 0x7FFFu + ((ub >> 16) & 1u)) >> 16;
  return (ua & 0xFFFFu) | (ub << 16);
}
static __device__ __forceinline__ __bf16 f2bf(float a) {
  unsigned ua = __builtin_bit_cast(unsigned, a);
  ua = (ua + 0x7FFFu + ((ua >> 16) & 1u)) >> 16;
  unsigned short s = (unsigned short)ua;
  return __builtin_bit_cast(__bf16, s);
}

// ------------------------------------------------------------------
// Kernel 0: weight pre-swizzle. Converts an [E][Ktot][Ntot] fp32
// tensor into bf16 WMMA B-fragments stored fragment-contiguous:
// tile (e, k0/32, n0/16) -> 256 u32; lane L owns u32[L*8 .. L*8+8).
// B layout used here (contiguous k, same as A loads): lanes 0-15 hold K=0..15 (u32 p =
// K pair 2p,2p+1), lanes 16-31 hold K=16..31; N = lane%16.
// One thread == one lane of one tile.
// ------------------------------------------------------------------
__global__ void swz_kernel(const float* __restrict__ src,
                           unsigned* __restrict__ dst,
                           int Ktot, int Ntot) {
  long long g = (long long)blockIdx.x * blockDim.x + threadIdx.x;
  int lane = (int)(g & 31);
  long long tile = g >> 5;
  int nTN = Ntot >> 4;
  int nTK = Ktot >> 5;
  long long tilesPerExp = (long long)nTK * nTN;
  int e = (int)(tile / tilesPerExp);
  long long rem = tile - (long long)e * tilesPerExp;
  int kt = (int)(rem / nTN);
  int nt = (int)(rem - (long long)kt * nTN);
  int lh = lane >> 4, ln = lane & 15;

  const float* s = src + ((size_t)e * Ktot + (size_t)kt * 32 + lh * 16) * Ntot
                       + (size_t)nt * 16 + ln;
  unsigned o[8];
#pragma unroll
  for (int p = 0; p < 8; ++p) {
    float b0 = s[(size_t)(2 * p) * Ntot];
    float b1 = s[(size_t)(2 * p + 1) * Ntot];
    o[p] = pk2bf(b0, b1);
  }
  unsigned* d = dst + (size_t)tile * 256 + (size_t)lane * 8;
  *(uint4*)(d + 0) = make_uint4(o[0], o[1], o[2], o[3]);
  *(uint4*)(d + 4) = make_uint4(o[4], o[5], o[6], o[7]);
}

// ------------------------------------------------------------------
// Kernel 1: zero the per-expert histogram + scatter cursors
// ------------------------------------------------------------------
__global__ void zero_kernel(int* p, int n) {
  int i = threadIdx.x;
  if (i < n) p[i] = 0;
}

// ------------------------------------------------------------------
// Kernel 2: router. One wave32 per token. logits = x[t] @ R (D x E),
// top-1 weight w = 1 / sum(exp(l - lmax)), idx = argmax.
// ------------------------------------------------------------------
__global__ void router_kernel(const float* __restrict__ x,
                              const float* __restrict__ R,
                              float* __restrict__ wout,
                              int* __restrict__ iout,
                              int* __restrict__ counts) {
  int wave = threadIdx.x >> 5;
  int lane = threadIdx.x & 31;
  int t = blockIdx.x * 8 + wave;
  const float* xr = x + (size_t)t * DIM;

  float acc[NEXP];
#pragma unroll
  for (int e = 0; e < NEXP; ++e) acc[e] = 0.f;

  for (int d = lane; d < DIM; d += 32) {
    float xv = xr[d];
    const float* rr = R + (size_t)d * NEXP;
#pragma unroll
    for (int e = 0; e < NEXP; ++e) acc[e] += xv * rr[e];
  }
#pragma unroll
  for (int off = 16; off > 0; off >>= 1) {
#pragma unroll
    for (int e = 0; e < NEXP; ++e) acc[e] += __shfl_xor(acc[e], off, 32);
  }
  float m = acc[0];
  int best = 0;
#pragma unroll
  for (int e = 1; e < NEXP; ++e) {
    if (acc[e] > m) { m = acc[e]; best = e; }   // strict > keeps first max (argmax)
  }
  float s = 0.f;
#pragma unroll
  for (int e = 0; e < NEXP; ++e) s += __expf(acc[e] - m);

  if (lane == 0) {
    wout[t] = 1.f / s;
    iout[t] = best;
    atomicAdd(&counts[best], 1);
  }
}

// ------------------------------------------------------------------
// Kernel 3: single-thread plan: exclusive prefix sum + tile list
// ------------------------------------------------------------------
__global__ void plan_kernel(const int* __restrict__ counts,
                            int* __restrict__ expOff,
                            int* __restrict__ tileE, int* __restrict__ tileS,
                            int* __restrict__ tileC, int* __restrict__ numTiles) {
  int off = 0, nt = 0;
  for (int e = 0; e < NEXP; ++e) {
    expOff[e] = off;
    int c = counts[e];
    for (int s = 0; s < c; s += TILE_M) {
      tileE[nt] = e;
      tileS[nt] = off + s;
      tileC[nt] = (c - s < TILE_M) ? (c - s) : TILE_M;
      ++nt;
    }
    off += c;
  }
  expOff[NEXP] = off;
  *numTiles = nt;
}

// ------------------------------------------------------------------
// Kernel 4: scatter tokens into per-expert buckets
// ------------------------------------------------------------------
__global__ void scatter_kernel(const int* __restrict__ idx,
                               const int* __restrict__ expOff,
                               int* __restrict__ cursor,
                               int* __restrict__ order) {
  int t = blockIdx.x * blockDim.x + threadIdx.x;
  int e = idx[t];
  int pos = expOff[e] + atomicAdd(&cursor[e], 1);
  order[pos] = t;
}

// ------------------------------------------------------------------
// Kernel 5: fused expert FFN for one 16-token tile.
//   h = silu(xtile @ W1_e), y = w[t] * (h @ W2_e)
// Weights arrive pre-swizzled bf16 (fragment-contiguous), so each
// B fragment is two global_load_b128 per lane; A fragments come
// from LDS. All math via v_wmma_f32_16x16x32_bf16.
// ------------------------------------------------------------------
#define XPAD 8    // 2064B row stride -> rows land on distinct LDS banks
#define HPAD 8

__global__ void __launch_bounds__(256, 1)
ffn_kernel(const float* __restrict__ xin,
           const unsigned* __restrict__ W1b,   // swizzled bf16 fragments
           const unsigned* __restrict__ W2b,   // swizzled bf16 fragments
           const float* __restrict__ wtok,
           const int* __restrict__ order,
           const int* __restrict__ tileE,
           const int* __restrict__ tileS,
           const int* __restrict__ tileC,
           const int* __restrict__ numTiles,
           float* __restrict__ out) {
  __shared__ __bf16 xtile[TILE_M][DIM + XPAD];
  __shared__ __bf16 htile[TILE_M][FFB + HPAD];
  __shared__ float  wrow[TILE_M];
  __shared__ int    trow[TILE_M];

  int tile = blockIdx.x;
  if (tile >= *numTiles) return;

  int e = tileE[tile];
  int start = tileS[tile];
  int cnt = tileC[tile];
  int tid = threadIdx.x;

  // ---- stage x tile into LDS as bf16 (16 rows x 16 loaders/row) ----
  {
    int r = tid >> 4, j = tid & 15;
    int t = (r < cnt) ? order[start + r] : -1;
    if (j == 0) { wrow[r] = (t >= 0) ? wtok[t] : 0.f; trow[r] = t; }
    unsigned* dst = (unsigned*)&xtile[r][0];
    if (t >= 0) {
      const float* src = xin + (size_t)t * DIM;
#pragma unroll
      for (int i = 0; i < 16; ++i) {
        int c = j * 64 + i * 4;
        float4 v = *(const float4*)(src + c);
        dst[(c >> 1) + 0] = pk2bf(v.x, v.y);
        dst[(c >> 1) + 1] = pk2bf(v.z, v.w);
      }
    } else {
#pragma unroll
      for (int i = 0; i < 16; ++i) {
        int c = j * 64 + i * 4;
        dst[(c >> 1) + 0] = 0u;
        dst[(c >> 1) + 1] = 0u;
      }
    }
  }
  __syncthreads();

  int wave = tid >> 5, lane = tid & 31;
  int lh = lane >> 4, ln = lane & 15;

  v8f acc[8];
#pragma unroll
  for (int i = 0; i < 8; ++i) acc[i] = (v8f)0.f;

#pragma unroll 1
  for (int ffb = 0; ffb < FFD; ffb += FFB) {
    // -------- GEMM1: 16x16 h-subtile at columns n0 = ffb + wave*16 --------
    int n0t = (ffb >> 4) + wave;                       // N-tile index into W1b
    v8f hacc = (v8f)0.f;
#pragma unroll 4
    for (int k0 = 0; k0 < DIM; k0 += 32) {
      union { v16bf v; unsigned u[8]; } A;
#pragma unroll
      for (int p = 0; p < 8; ++p) {
        int K = k0 + lh * 16 + 2 * p;
        A.u[p] = *(const unsigned*)&xtile[ln][K];
      }
      size_t tIdx = ((size_t)e * (DIM / 32) + (k0 >> 5)) * (FFD / 16) + n0t;
      const uint4* bp = (const uint4*)(W1b + tIdx * 256 + (size_t)lane * 8);
      union { v16bf v; uint4 q[2]; } Bf;
      Bf.q[0] = bp[0];
      Bf.q[1] = bp[1];
      hacc = __builtin_amdgcn_wmma_f32_16x16x32_bf16(
          false, A.v, false, Bf.v, (short)0, hacc, false, false);
    }
    // silu -> htile (bf16); fast path: v_exp + v_rcp
#pragma unroll
    for (int v = 0; v < 8; ++v) {
      int m = lh * 8 + v;
      float hx = hacc[v];
      float sv = hx * __builtin_amdgcn_rcpf(1.f + __expf(-hx));
      htile[m][wave * 16 + ln] = f2bf(sv);
    }
    __syncthreads();

    // -------- GEMM2: out[16, 128w .. 128w+128) += h @ W2 --------
#pragma unroll 1
    for (int kk = 0; kk < FFB; kk += 32) {
      union { v16bf v; unsigned u[8]; } A2;
#pragma unroll
      for (int p = 0; p < 8; ++p) {
        int K = kk + lh * 16 + 2 * p;
        A2.u[p] = *(const unsigned*)&htile[ln][K];
      }
      size_t tBase = ((size_t)e * (FFD / 32) + ((ffb + kk) >> 5)) * (DIM / 16)
                   + (size_t)wave * 8;
#pragma unroll
      for (int nt = 0; nt < 8; ++nt) {
        const uint4* bp = (const uint4*)(W2b + (tBase + nt) * 256 + (size_t)lane * 8);
        union { v16bf v; uint4 q[2]; } B2;
        B2.q[0] = bp[0];
        B2.q[1] = bp[1];
        acc[nt] = __builtin_amdgcn_wmma_f32_16x16x32_bf16(
            false, A2.v, false, B2.v, (short)0, acc[nt], false, false);
      }
    }
    __syncthreads();
  }

  // ---- writeback: y[t] = w[t] * acc ----
#pragma unroll
  for (int v = 0; v < 8; ++v) {
    int m = lh * 8 + v;
    int t = trow[m];
    if (t >= 0) {
      float sc = wrow[m];
      float* op = out + (size_t)t * DIM + wave * 128 + ln;
#pragma unroll
      for (int nt = 0; nt < 8; ++nt) op[nt * 16] = acc[nt][v] * sc;
    }
  }
}

// ------------------------------------------------------------------
// Host launcher.
//   pre-pass: swizzle W1, W2 -> bf16 fragment buffers (once per call)
//   4 layers, activations ping-pong ws <-> d_out:
//   l0: x -> ws ; l1: ws -> out ; l2: out -> ws ; l3: ws -> out
// ------------------------------------------------------------------
extern "C" void kernel_launch(void* const* d_in, const int* in_sizes, int n_in,
                              void* d_out, int out_size, void* d_ws, size_t ws_size,
                              hipStream_t stream) {
  (void)in_sizes; (void)n_in; (void)out_size; (void)ws_size;

  const float* x       = (const float*)d_in[0];   // [B,S,D]
  const float* routers = (const float*)d_in[1];   // [L,D,E]
  const float* W1      = (const float*)d_in[2];   // [E,D,FF]
  const float* W2      = (const float*)d_in[3];   // [E,FF,D]
  float* out = (float*)d_out;

  char* ws = (char*)d_ws;
  size_t off = 0;
  float*    X1   = (float*)(ws + off);    off += (size_t)T_TOK * DIM * sizeof(float);
  unsigned* W1b  = (unsigned*)(ws + off); off += (size_t)NEXP * DIM * FFD / 2 * sizeof(unsigned);
  unsigned* W2b  = (unsigned*)(ws + off); off += (size_t)NEXP * FFD * DIM / 2 * sizeof(unsigned);
  float* wtok    = (float*)(ws + off);    off += (size_t)T_TOK * sizeof(float);
  int*   idx     = (int*)(ws + off);      off += (size_t)T_TOK * sizeof(int);
  int*   order   = (int*)(ws + off);      off += (size_t)T_TOK * sizeof(int);
  int*   counts  = (int*)(ws + off);      off += 16 * sizeof(int);
  int*   cursor  = (int*)(ws + off);      off += 16 * sizeof(int);
  int*   expOff  = (int*)(ws + off);      off += 32 * sizeof(int);  // 17 used, pad
  int*   tileE   = (int*)(ws + off);      off += MAX_TILES * sizeof(int);
  int*   tileS   = (int*)(ws + off);      off += MAX_TILES * sizeof(int);
  int*   tileC   = (int*)(ws + off);      off += MAX_TILES * sizeof(int);
  int*   numT    = (int*)(ws + off);      off += sizeof(int);

  // ---- one-time (per call) weight swizzle: fp32 -> bf16 B-fragments ----
  {
    long long thrW1 = (long long)NEXP * (DIM / 32) * (FFD / 16) * 32;   // 4,194,304
    long long thrW2 = (long long)NEXP * (FFD / 32) * (DIM / 16) * 32;   // 4,194,304
    swz_kernel<<<(unsigned)(thrW1 / 256), 256, 0, stream>>>(W1, W1b, DIM, FFD);
    swz_kernel<<<(unsigned)(thrW2 / 256), 256, 0, stream>>>(W2, W2b, FFD, DIM);
  }

  const float* bufs_in[NLAY]  = { x,  X1,  out, X1 };
  float*       bufs_out[NLAY] = { X1, out, X1,  out };

  for (int l = 0; l < NLAY; ++l) {
    zero_kernel<<<1, 64, 0, stream>>>(counts, 32);  // counts + cursor contiguous
    router_kernel<<<T_TOK / 8, 256, 0, stream>>>(
        bufs_in[l], routers + (size_t)l * DIM * NEXP, wtok, idx, counts);
    plan_kernel<<<1, 1, 0, stream>>>(counts, expOff, tileE, tileS, tileC, numT);
    scatter_kernel<<<T_TOK / 256, 256, 0, stream>>>(idx, expOff, cursor, order);
    ffn_kernel<<<MAX_TILES, 256, 0, stream>>>(
        bufs_in[l], W1b, W2b, wtok, order, tileE, tileS, tileC, numT, bufs_out[l]);
  }
}
